// RecentAttention_62294205661438
// MI455X (gfx1250) — matrix operand order repaired and checked by reading, not compiled, not run on hardware
//
#include <hip/hip_runtime.h>
#include <math.h>

typedef __attribute__((ext_vector_type(16))) _Float16 v16h;
typedef __attribute__((ext_vector_type(8)))  float    v8f;
typedef __attribute__((ext_vector_type(4)))  unsigned int u32x4;
typedef __attribute__((ext_vector_type(8)))  int      i32x8;
typedef __attribute__((ext_vector_type(4)))  int      i32x4;

#define H 128
#define XS_STRIDE 132   // floats/row: 528B (16B aligned), bank stride 4 -> 16 rows distinct banks
#define WH_STRIDE 136   // halfs/row: 272B (16B aligned), bank stride 4 -> 16 lanes distinct banks

#if __has_builtin(__builtin_amdgcn_tensor_load_to_lds) && \
    __has_builtin(__builtin_amdgcn_s_wait_tensorcnt)
#define USE_TDM 1
#else
#define USE_TDM 0
#endif

#if USE_TDM
// Issue one 2-D TDM tile load: global (row-major, stride_d0 elems) -> LDS with
// optional per-row padding. D# layout per CDNA5 ISA §8.3/8.4.
// This toolchain's builtin is the 6-arg form:
//   (u32x4 g0, i32x8 g1, i32x4 g2, i32x4 g3, i32x8 extra, i32 cpol)
__device__ __forceinline__ void tdm_load_2d(unsigned lds_off, const void* gptr,
                                            unsigned elem_code,      // 0=1B,1=2B,2=4B
                                            unsigned tensor_d0, unsigned tensor_d1,
                                            unsigned tile_d0, unsigned tile_d1,
                                            unsigned stride_d0,
                                            unsigned pad_interval_code,
                                            unsigned pad_amount_code,
                                            unsigned pad_enable) {
  unsigned long long ga = (unsigned long long)gptr;
  u32x4 g0;
  g0[0] = 1u;                                   // count=1, user descriptor
  g0[1] = lds_off;                              // lds_addr [63:32]
  g0[2] = (unsigned)ga;                         // global_addr [95:64]
  g0[3] = (unsigned)((ga >> 32) & 0x1FFFFFFu) | 0x80000000u;  // addr[120:96] | type=2
  i32x8 g1;
  g1[0] = (int)((elem_code << 16) | (pad_enable << 20) |
                (pad_interval_code << 22) | (pad_amount_code << 25));
  g1[1] = (int)((tensor_d0 & 0xFFFFu) << 16);   // tensor_dim0 [79:48] low
  g1[2] = (int)((tensor_d0 >> 16) | ((tensor_d1 & 0xFFFFu) << 16));
  g1[3] = (int)(((tensor_d1 >> 16) & 0xFFFFu) | ((tile_d0 & 0xFFFFu) << 16));
  g1[4] = (int)(tile_d1 & 0xFFFFu);             // tile_dim1; tile_dim2=0
  g1[5] = (int)stride_d0;                       // tensor_dim0_stride [191:160]
  g1[6] = 0;                                    // stride high / dim1_stride low
  g1[7] = 0;
  i32x4 gz4 = {0, 0, 0, 0};
  i32x8 gz8 = {0, 0, 0, 0, 0, 0, 0, 0};
  __builtin_amdgcn_tensor_load_to_lds(g0, g1, gz4, gz4, gz8, 0);
}
#endif

// ---------------------------------------------------------------------------
// Kernel 0: one-time f32 -> f16 conversion of W1 and W2 into workspace
// ---------------------------------------------------------------------------
__global__ __launch_bounds__(256)
void cvt_weights(const float* __restrict__ W1, const float* __restrict__ W2,
                 _Float16* __restrict__ W1h, _Float16* __restrict__ W2h) {
  int i = blockIdx.x * blockDim.x + threadIdx.x;
  if (i < H * H) {
    W1h[i] = (_Float16)W1[i];
    W2h[i] = (_Float16)W2[i];
  }
}

// ---------------------------------------------------------------------------
// Kernel 1: c[b][j] = dot(x[last_ixs[b]], W1[j]) + W1_b[j] + W2_b[j]
// ---------------------------------------------------------------------------
__global__ __launch_bounds__(256)
void session_bias_kernel(const float* __restrict__ x,
                         const _Float16* __restrict__ W1h,
                         const float* __restrict__ W1b,
                         const float* __restrict__ W2b,
                         const int* __restrict__ last_ixs,
                         float* __restrict__ cfull, int B) {
  __shared__ __align__(16) float    vl[64 * XS_STRIDE];
  __shared__ __align__(16) _Float16 w1l[H * WH_STRIDE];
  __shared__ float bsum[H];

  const int tid = threadIdx.x;
  const int blk = blockIdx.x;

  // gather last-row of each session (128-bit loads; pad with zeros past B)
  const float4* xg4 = (const float4*)x;
  for (int idx = tid; idx < 64 * 32; idx += 256) {
    int m = idx >> 5, q = idx & 31;
    int sess = blk * 64 + m;
    float4 v = make_float4(0.f, 0.f, 0.f, 0.f);
    if (sess < B) v = xg4[(long)last_ixs[sess] * 32 + q];
    *(float4*)&vl[m * XS_STRIDE + q * 4] = v;
  }
#if USE_TDM
  if (tid < 32) {
    // W1h f16 [128][128] -> LDS rows padded to WH_STRIDE halfs.
    // tile row = 128 halfs = 64 dwords (interval code 5), pad 4 dwords (code 3).
    tdm_load_2d((unsigned)(unsigned long long)(void*)w1l, (const void*)W1h,
                1u, H, H, H, H, H, 5u, 3u, 1u);
    __builtin_amdgcn_s_wait_tensorcnt(0);
  }
#else
  {
    const uint4* wg4 = (const uint4*)W1h;
    for (int idx = tid; idx < H * 16; idx += 256) {   // 16 uint4 per 128-half row
      int j = idx >> 4, q = idx & 15;
      *(uint4*)&w1l[j * WH_STRIDE + q * 8] = wg4[j * 16 + q];
    }
  }
#endif
  if (tid < H) bsum[tid] = W1b[tid] + W2b[tid];
  __syncthreads();

  const int lane   = tid & 31;
  const int w      = tid >> 5;
  const int mt     = w & 3;
  const int nbase  = (w >> 2) * 64;
  const int mrow   = mt * 16 + (lane & 15);
  const int khalfA = (lane >= 16) ? 16 : 0;
  const int khalfB = (lane >= 16) ? 16 : 0;
  const int ncol   = lane & 15;

  v8f acc[4];
#pragma unroll
  for (int nt = 0; nt < 4; ++nt) acc[nt] = (v8f){0, 0, 0, 0, 0, 0, 0, 0};

#pragma unroll
  for (int k0 = 0; k0 < H; k0 += 32) {
    v16h a;
#pragma unroll
    for (int t = 0; t < 16; ++t) {
      int K = k0 + khalfA + t;
      a[t] = (_Float16)vl[mrow * XS_STRIDE + K];
    }
#pragma unroll
    for (int nt = 0; nt < 4; ++nt) {
      int j = nbase + nt * 16 + ncol;
      const _Float16* bp = &w1l[j * WH_STRIDE + k0 + khalfB];
      v16h b;
#pragma unroll
      for (int t = 0; t < 16; ++t) b[t] = bp[t];
      acc[nt] = __builtin_amdgcn_wmma_f32_16x16x32_f16(
          false, a, false, b, (short)0, acc[nt], false, false);
    }
  }

  // store: fast path for full blocks (no per-element guards)
  const int mlocal = mt * 16 + ((lane >= 16) ? 8 : 0);
  float* op = cfull + ((long)(blk * 64 + mlocal)) * H + nbase + ncol;
  float bj[4];
#pragma unroll
  for (int nt = 0; nt < 4; ++nt) bj[nt] = bsum[nbase + nt * 16 + ncol];
  if (blk * 64 + 64 <= B) {
#pragma unroll
    for (int v = 0; v < 8; ++v)
#pragma unroll
      for (int nt = 0; nt < 4; ++nt)
        op[(long)v * H + nt * 16] = acc[nt][v] + bj[nt];
  } else {
#pragma unroll
    for (int v = 0; v < 8; ++v)
      if (blk * 64 + mlocal + v < B)
#pragma unroll
        for (int nt = 0; nt < 4; ++nt)
          op[(long)v * H + nt * 16] = acc[nt][v] + bj[nt];
  }
}

// ---------------------------------------------------------------------------
// Kernel 2: per-session: h = sigmoid(x@W2^T + c), e = h.q + qb,
// per-session softmax, out = sum alpha*x.  (L <= 64; here L = 50.)
// ---------------------------------------------------------------------------
__global__ __launch_bounds__(256)
void session_attn_kernel(const float* __restrict__ x,
                         const _Float16* __restrict__ W2h,
                         const float* __restrict__ cfull,
                         const float* __restrict__ qw,
                         const float* __restrict__ qb,
                         float* __restrict__ out, int L) {
  __shared__ __align__(16) float    xs[64 * XS_STRIDE];
  __shared__ __align__(16) _Float16 w2l[H * WH_STRIDE];
  __shared__ float cl[H];
  __shared__ float ql[H];
  __shared__ float e_l[64];
  __shared__ float z_l[64];
  __shared__ float pool[H];
  __shared__ float sinv_l;

  const int tid = threadIdx.x;
  const int sid = blockIdx.x;
  const float qb0 = qb[0];

#if USE_TDM
  if (tid < 32) {
    // x tile [L][128] f32 -> xs rows padded to XS_STRIDE floats.
    // tile row = 128 dwords (interval code 6), pad 4 dwords (code 3).
    tdm_load_2d((unsigned)(unsigned long long)(void*)xs,
                (const void*)(x + (long)sid * L * H),
                2u, H, (unsigned)L, H, (unsigned)L, H, 6u, 3u, 1u);
    // W2 f16 tile, same shape as kernel 1.
    tdm_load_2d((unsigned)(unsigned long long)(void*)w2l, (const void*)W2h,
                1u, H, H, H, H, H, 5u, 3u, 1u);
    __builtin_amdgcn_s_wait_tensorcnt(0);
  }
  // zero the padding rows L..63 (disjoint from TDM destination rows)
  for (int idx = tid; idx < (64 - L) * 32; idx += 256) {
    int m = L + (idx >> 5), q = idx & 31;
    *(float4*)&xs[m * XS_STRIDE + q * 4] = make_float4(0.f, 0.f, 0.f, 0.f);
  }
#else
  {
    const float4* xg4 = (const float4*)x;
    for (int idx = tid; idx < 64 * 32; idx += 256) {
      int m = idx >> 5, q = idx & 31;
      float4 v = make_float4(0.f, 0.f, 0.f, 0.f);
      if (m < L) v = xg4[((long)sid * L + m) * 32 + q];
      *(float4*)&xs[m * XS_STRIDE + q * 4] = v;
    }
    const uint4* wg4 = (const uint4*)W2h;
    for (int idx = tid; idx < H * 16; idx += 256) {
      int j = idx >> 4, q = idx & 15;
      *(uint4*)&w2l[j * WH_STRIDE + q * 8] = wg4[j * 16 + q];
    }
  }
#endif
  if (tid < H) { cl[tid] = cfull[(long)sid * H + tid]; ql[tid] = qw[tid]; }
  if (tid < 64) e_l[tid] = qb0;
  __syncthreads();

  const int lane   = tid & 31;
  const int w      = tid >> 5;
  const int mt     = w & 3;
  const int nbase  = (w >> 2) * 64;
  const int mrow   = mt * 16 + (lane & 15);
  const int khalfA = (lane >= 16) ? 16 : 0;
  const int khalfB = (lane >= 16) ? 16 : 0;
  const int ncol   = lane & 15;

  v8f acc[4];
#pragma unroll
  for (int nt = 0; nt < 4; ++nt) acc[nt] = (v8f){0, 0, 0, 0, 0, 0, 0, 0};

#pragma unroll
  for (int k0 = 0; k0 < H; k0 += 32) {
    v16h a;
#pragma unroll
    for (int t = 0; t < 16; ++t) {
      int K = k0 + khalfA + t;
      a[t] = (_Float16)xs[mrow * XS_STRIDE + K];
    }
#pragma unroll
    for (int nt = 0; nt < 4; ++nt) {
      int j = nbase + nt * 16 + ncol;
      const _Float16* bp = &w2l[j * WH_STRIDE + k0 + khalfB];
      v16h b;
#pragma unroll
      for (int t = 0; t < 16; ++t) b[t] = bp[t];
      acc[nt] = __builtin_amdgcn_wmma_f32_16x16x32_f16(
          false, a, false, b, (short)0, acc[nt], false, false);
    }
  }

  // epilogue: p[v] = sum over wave's 64 cols of sigmoid(d + c[j]) * q[j]
  float p[8] = {0, 0, 0, 0, 0, 0, 0, 0};
#pragma unroll
  for (int nt = 0; nt < 4; ++nt) {
    int j = nbase + nt * 16 + ncol;
    float cj = cl[j];
    float qj = ql[j];
#pragma unroll
    for (int v = 0; v < 8; ++v) {
      float t = acc[nt][v] + cj;
      float hv = 1.0f / (1.0f + expf(-t));
      p[v] += hv * qj;
    }
  }
#pragma unroll
  for (int v = 0; v < 8; ++v) {
#pragma unroll
    for (int mask = 1; mask <= 8; mask <<= 1)
      p[v] += __shfl_xor(p[v], mask, 32);
  }
  if ((lane & 15) == 0) {
    int mbase = mt * 16 + ((lane >= 16) ? 8 : 0);
#pragma unroll
    for (int v = 0; v < 8; ++v) atomicAdd(&e_l[mbase + v], p[v]);
  }
  __syncthreads();

  // wave 0: numerically stable softmax over e_l[0..L)
  if (tid < 32) {
    float e1 = (tid < L) ? e_l[tid] : -1e30f;
    float e2 = (tid + 32 < L) ? e_l[tid + 32] : -1e30f;
    float vmax = fmaxf(e1, e2);
#pragma unroll
    for (int mask = 16; mask >= 1; mask >>= 1)
      vmax = fmaxf(vmax, __shfl_xor(vmax, mask, 32));
    float z1 = (tid < L) ? expf(e1 - vmax) : 0.0f;
    float z2 = (tid + 32 < L) ? expf(e2 - vmax) : 0.0f;
    z_l[tid] = z1;
    z_l[tid + 32] = z2;
    float s = z1 + z2;
#pragma unroll
    for (int mask = 16; mask >= 1; mask >>= 1)
      s += __shfl_xor(s, mask, 32);
    if (tid == 0) sinv_l = 1.0f / s;
  }
  __syncthreads();

  // pooling: out[sid][j] = sum_m alpha_m * x[m][j]  (fp32 from LDS copy)
  const float si = sinv_l;
  const int j = tid & 127;
  const int half = tid >> 7;
  const int mmid = (L + 1) >> 1;
  const int mstart = half ? mmid : 0;
  const int mend   = half ? L : mmid;
  float partial = 0.0f;
  for (int m = mstart; m < mend; ++m)
    partial += z_l[m] * xs[m * XS_STRIDE + j];
  if (half) pool[j] = partial;
  __syncthreads();
  if (!half) out[(long)sid * H + j] = (partial + pool[j]) * si;
}

// ---------------------------------------------------------------------------
extern "C" void kernel_launch(void* const* d_in, const int* in_sizes, int n_in,
                              void* d_out, int out_size, void* d_ws, size_t ws_size,
                              hipStream_t stream) {
  const float* x    = (const float*)d_in[0];
  const float* W1w  = (const float*)d_in[1];
  const float* W1b  = (const float*)d_in[2];
  const float* W2w  = (const float*)d_in[3];
  const float* W2b  = (const float*)d_in[4];
  const float* qw   = (const float*)d_in[5];
  const float* qb   = (const float*)d_in[6];
  // d_in[7] = batch (segment ids) — implied by contiguous equal-length layout
  const int* last_ixs = (const int*)d_in[8];

  const int B = in_sizes[8];          // number of sessions
  const int Ntok = in_sizes[0] / H;   // total rows
  const int L = Ntok / B;             // rows per session (<= 64)

  char* ws = (char*)d_ws;
  _Float16* W1h  = (_Float16*)ws;                 // 32 KB
  _Float16* W2h  = (_Float16*)(ws + 32768);       // 32 KB
  float*    cful = (float*)(ws + 65536);          // B*128 f32

  cvt_weights<<<(H * H + 255) / 256, 256, 0, stream>>>(W1w, W2w, W1h, W2h);
  session_bias_kernel<<<(B + 63) / 64, 256, 0, stream>>>(x, W1h, W1b, W2b,
                                                         last_ixs, cful, B);
  session_attn_kernel<<<B, 256, 0, stream>>>(x, W2h, cful, qw, qb,
                                             (float*)d_out, L);
}
